// MultiHeadAttention_9380208574720
// MI455X (gfx1250) — compile-verified, matrix operand order checked by reading, not run on hardware
//
#include <hip/hip_runtime.h>
#include <hip/hip_bf16.h>
#include <math.h>

typedef __attribute__((ext_vector_type(16))) _Float16     v16h;
typedef __attribute__((ext_vector_type(8)))  float        v8f;
typedef __attribute__((ext_vector_type(4)))  unsigned int u32x4;
typedef __attribute__((ext_vector_type(8)))  int          i32x8;
typedef __attribute__((ext_vector_type(4)))  int          i32x4;

#define B_  2
#define S_  2048
#define D_  1024
#define H_  16
#define DK_ 64

#if defined(__has_builtin)
#if __has_builtin(__builtin_amdgcn_tensor_load_to_lds) && __has_builtin(__builtin_amdgcn_s_wait_tensorcnt)
#define HAVE_TDM 1
#endif
#endif

union V16U { uint4 u[2]; v16h h; };
__device__ inline v16h ld16h_lds(const _Float16* p) {
    V16U t;
    t.u[0] = *(const uint4*)p;
    t.u[1] = *(const uint4*)(p + 8);
    return t.h;
}

#ifdef HAVE_TDM
// Issue one 2-D TDM tile load: tile_rows x 32 f16 elements from a row-major
// [tensor_rows x 1024] f16 tensor into LDS, padded +4 DWORDs every 16 DWORDs
// (so LDS row stride = 40 halfs). D# packing per CDNA5 ISA 8.3/8.4.
// This toolchain's builtin takes 6 args: (g0, g1, g2, g3, extra, cpol).
__device__ inline void tdm_load_tile(unsigned lds_addr, const _Float16* gptr,
                                     int tile_rows, int tensor_rows) {
    const unsigned long long ga = (unsigned long long)(size_t)gptr;
    u32x4 g0;
    g0[0] = 1u;                                            // count=1, user desc
    g0[1] = lds_addr;                                      // lds_addr [63:32]
    g0[2] = (unsigned)ga;                                  // global_addr lo
    g0[3] = ((unsigned)(ga >> 32) & 0x01FFFFFFu)           // global_addr hi
          | 0x80000000u;                                   // type=2 ("image")
    i32x8 g1;
    g1[0] = (1 << 16)                                      // data_size = 2B
          | (1 << 20)                                      // pad_enable
          | (3 << 22)                                      // pad_interval: 16 DW
          | (3 << 25);                                     // pad_amount: 4 DW
    g1[1] = (int)((D_ & 0xFFFF) << 16);                    // tensor_dim0 lo16
    g1[2] = (int)((tensor_rows & 0xFFFF) << 16);           // tensor_dim1 lo16
    g1[3] = (int)(32u << 16);                              // tile_dim0 = 32
    g1[4] = tile_rows;                                     // tile_dim1 (dim2=0)
    g1[5] = D_;                                            // tensor_dim0_stride
    g1[6] = 0;
    g1[7] = 0;
    const i32x4 z4 = {0, 0, 0, 0};                         // 2-D: groups 2/3 off
    const i32x8 z8 = {0, 0, 0, 0, 0, 0, 0, 0};
    __builtin_amdgcn_tensor_load_to_lds(g0, g1, z4, z4, z8, 0);
}
#endif

// ---------------------------------------------------------------------------
// f32 -> f16 convert (vectorized, n multiple of 2048)
// ---------------------------------------------------------------------------
__global__ __launch_bounds__(256) void cvt_f32_f16(const float* __restrict__ s,
                                                   _Float16* __restrict__ d) {
    const int i = (blockIdx.x * 256 + threadIdx.x) * 8;
    const float4 a = *(const float4*)(s + i);
    const float4 b = *(const float4*)(s + i + 4);
    union { _Float16 h[8]; uint4 u; } t;
    t.h[0] = (_Float16)a.x; t.h[1] = (_Float16)a.y;
    t.h[2] = (_Float16)a.z; t.h[3] = (_Float16)a.w;
    t.h[4] = (_Float16)b.x; t.h[5] = (_Float16)b.y;
    t.h[6] = (_Float16)b.z; t.h[7] = (_Float16)b.w;
    *(uint4*)(d + i) = t.u;
}

// ---------------------------------------------------------------------------
// GEMM: Y[m,n] = sum_k A[m,k] * W[n,k] + bias[n]   (A,W already f16)
// MODE 0: out=f16 head layout  [b][h][s][dk]
// MODE 1: out=f16 transposed-V [b][h][dk][s]
// MODE 2: out=f32 row-major    [m][n]
// 256 threads (8 waves). Block tile 128(M) x 64(N); k-chunks of 32.
// Tile staging by the Tensor Data Mover (wave0 issues 2 descriptors per
// chunk, double-buffered LDS, s_wait_tensorcnt + one barrier per chunk).
// ---------------------------------------------------------------------------
template<int MODE>
__global__ __launch_bounds__(256) void gemm_wmma(const _Float16* __restrict__ Ain,
                                                 const _Float16* __restrict__ W,
                                                 const float* __restrict__ bias,
                                                 void* __restrict__ out)
{
    constexpr int LDA = 40;                       // halfs per LDS row (80B)
    constexpr int NK  = D_ / 32;                  // 32 k-chunks
    __shared__ _Float16 sA[2][128 * LDA];
    __shared__ _Float16 sW[2][64 * LDA];

    const int t      = threadIdx.x;
    const int wv     = t >> 5;
    const int lane   = t & 31;
    const int lanelo = lane & 15;
    const int lanehi = lane >> 4;
    const int m0     = blockIdx.x * 128;
    const int n0     = blockIdx.y * 64;

    v8f acc[4] = {};

#ifdef HAVE_TDM
    // ---- prologue: DMA chunk 0 into buffer 0 ----
    if (wv == 0) {
        tdm_load_tile((unsigned)(size_t)&sA[0][0], Ain + (size_t)m0 * D_, 128, 4096);
        tdm_load_tile((unsigned)(size_t)&sW[0][0], W   + (size_t)n0 * D_, 64, 1024);
    }
    for (int kc = 0; kc < NK; ++kc) {
        const int cur = kc & 1;
        __builtin_amdgcn_s_wait_tensorcnt(0);     // issuer drains its DMAs
        __syncthreads();                          // everyone sees buffer `cur`
        if (wv == 0 && kc + 1 < NK) {             // DMA next chunk -> other buf
            const int k0n = (kc + 1) * 32;
            tdm_load_tile((unsigned)(size_t)&sA[cur ^ 1][0],
                          Ain + (size_t)m0 * D_ + k0n, 128, 4096);
            tdm_load_tile((unsigned)(size_t)&sW[cur ^ 1][0],
                          W + (size_t)n0 * D_ + k0n, 64, 1024);
        }
        const v16h af = ld16h_lds(&sA[cur][(wv * 16 + lanelo) * LDA + 16 * lanehi]);
        #pragma unroll
        for (int j = 0; j < 4; ++j) {
            const v16h bf = ld16h_lds(&sW[cur][(j * 16 + lanelo) * LDA + 16 * lanehi]);
            acc[j] = __builtin_amdgcn_wmma_f32_16x16x32_f16(
                         false, af, false, bf, (short)0, acc[j], false, false);
        }
    }
#else
    // ---- fallback: cooperative copy staging (single buffer) ----
    for (int kc = 0; kc < NK; ++kc) {
        const int k0 = kc * 32;
        #pragma unroll
        for (int j = 0; j < 2; ++j) {             // A tile 128x32 halfs
            const int e = 8 * t + 2048 * j, r = e >> 5, c = e & 31;
            *(uint4*)(&sA[0][r * LDA + c]) =
                *(const uint4*)(Ain + (size_t)(m0 + r) * D_ + k0 + c);
        }
        {                                         // W tile 64x32 halfs
            const int e = 8 * t, r = e >> 5, c = e & 31;
            if (r < 64)
                *(uint4*)(&sW[0][r * LDA + c]) =
                    *(const uint4*)(W + (size_t)(n0 + r) * D_ + k0 + c);
        }
        __syncthreads();
        const v16h af = ld16h_lds(&sA[0][(wv * 16 + lanelo) * LDA + 16 * lanehi]);
        #pragma unroll
        for (int j = 0; j < 4; ++j) {
            const v16h bf = ld16h_lds(&sW[0][(j * 16 + lanelo) * LDA + 16 * lanehi]);
            acc[j] = __builtin_amdgcn_wmma_f32_16x16x32_f16(
                         false, af, false, bf, (short)0, acc[j], false, false);
        }
        __syncthreads();
    }
#endif

    // ---- epilogue: bias + store ----
    #pragma unroll
    for (int j = 0; j < 4; ++j) {
        const int col = n0 + j * 16 + lanelo;
        const float bn = bias[col];
        #pragma unroll
        for (int r = 0; r < 8; ++r) {
            const int m = m0 + wv * 16 + r + 8 * lanehi;
            const float y = acc[j][r] + bn;
            if (MODE == 2) {
                ((float*)out)[(size_t)m * D_ + col] = y;
            } else {
                const int b = m >> 11, s = m & (S_ - 1);
                const int h = col >> 6, dk = col & 63;
                _Float16* o16 = (_Float16*)out;
                if (MODE == 0)
                    o16[((size_t)(b * H_ + h) * S_ + s) * DK_ + dk] = (_Float16)y;
                else
                    o16[((size_t)(b * H_ + h) * DK_ + dk) * S_ + s] = (_Float16)y;
            }
        }
    }
}

// ---------------------------------------------------------------------------
// Flash attention: wave = 16 query rows of one (b,h); 64-key tiles.
// Per tile: 8 score WMMAs + 8 PV WMMAs + 2 row-sum WMMAs (P x ones).
// Block = 8 waves = 128 query rows. Grid = (S/128, B*H).
// ---------------------------------------------------------------------------
__global__ __launch_bounds__(256) void attn_wmma(const _Float16* __restrict__ qh,
                                                 const _Float16* __restrict__ kh,
                                                 const _Float16* __restrict__ vT,
                                                 _Float16* __restrict__ O)
{
    constexpr int LDP = 72;                       // halfs per LDS row (144B)
    __shared__ _Float16 sP[8 * 16 * LDP];         // per-wave private 16x64 P tile

    const int t      = threadIdx.x;
    const int wv     = t >> 5;
    const int lane   = t & 31;
    const int lanelo = lane & 15;
    const int lanehi = lane >> 4;
    const int bh     = blockIdx.y;
    const int q0     = blockIdx.x * 128 + wv * 16;

    const _Float16* qb = qh + (size_t)bh * S_ * DK_;
    const _Float16* kb = kh + (size_t)bh * S_ * DK_;
    const _Float16* vb = vT + (size_t)bh * DK_ * S_;
    _Float16*      sPw = &sP[wv * 16 * LDP];

    // Q fragments with 1/sqrt(64) pre-folded (exact in f16)
    v16h qa0 = *(const v16h*)(qb + (size_t)(q0 + lanelo) * DK_ + 16 * lanehi);
    v16h qa1 = *(const v16h*)(qb + (size_t)(q0 + lanelo) * DK_ + 32 + 16 * lanehi);
    qa0 *= (_Float16)0.125f;
    qa1 *= (_Float16)0.125f;

    v16h ones;
    #pragma unroll
    for (int i = 0; i < 16; ++i) ones[i] = (_Float16)1.0f;

    v8f o[4] = {};                                // 16x64 output accumulator
    v8f osum = {};                                // 16x16 row-sum accumulator
    float mrun[8];
    #pragma unroll
    for (int r = 0; r < 8; ++r) mrun[r] = -3.0e38f;

    for (int kt = 0; kt < S_; kt += 64) {
        // ---- scores: 16 queries x 64 keys = four C tiles, 8 WMMAs ----
        v8f s[4];
        #pragma unroll
        for (int kk = 0; kk < 4; ++kk) {
            const _Float16* krow = kb + (size_t)(kt + kk * 16 + lanelo) * DK_;
            const v16h k0 = *(const v16h*)(krow + 16 * lanehi);
            const v16h k1 = *(const v16h*)(krow + 32 + 16 * lanehi);
            v8f z = {};
            z = __builtin_amdgcn_wmma_f32_16x16x32_f16(false, qa0, false, k0, (short)0, z, false, false);
            s[kk] = __builtin_amdgcn_wmma_f32_16x16x32_f16(false, qa1, false, k1, (short)0, z, false, false);
        }
        if (kt + 64 < S_) {
            __builtin_prefetch(kb + (size_t)(kt + 64 + lanelo) * DK_, 0, 3);
            __builtin_prefetch(vb + (size_t)lanelo * S_ + kt + 64, 0, 3);
        }

        // ---- running max + exp; stage P into wave-private LDS ----
        #pragma unroll
        for (int r = 0; r < 8; ++r) {
            float rmax = fmaxf(fmaxf(s[0][r], s[1][r]), fmaxf(s[2][r], s[3][r]));
            #pragma unroll
            for (int mk = 1; mk < 16; mk <<= 1)
                rmax = fmaxf(rmax, __shfl_xor(rmax, mk, 32));
            const float mnew  = fmaxf(mrun[r], rmax);
            const float alpha = __expf(mrun[r] - mnew);
            mrun[r] = mnew;

            const int pm = r + 8 * lanehi;
            #pragma unroll
            for (int kk = 0; kk < 4; ++kk)
                sPw[pm * LDP + kk * 16 + lanelo] = (_Float16)__expf(s[kk][r] - mnew);

            #pragma unroll
            for (int j = 0; j < 4; ++j) o[j][r] *= alpha;
            osum[r] *= alpha;
        }

        // intra-wave LDS store->load ordering (in-order LDS pipe + drain DScnt)
        asm volatile("s_wait_dscnt 0x0" ::: "memory");

        // P as two 16x32 A-fragments
        const v16h pa0 = ld16h_lds(&sPw[lanelo * LDP + 16 * lanehi]);
        const v16h pa1 = ld16h_lds(&sPw[lanelo * LDP + 32 + 16 * lanehi]);

        // ---- row-sum on the matrix pipe: osum += P x ones ----
        osum = __builtin_amdgcn_wmma_f32_16x16x32_f16(false, pa0, false, ones, (short)0, osum, false, false);
        osum = __builtin_amdgcn_wmma_f32_16x16x32_f16(false, pa1, false, ones, (short)0, osum, false, false);

        // ---- O += P x V : 8 WMMAs over DK=64 (V stored transposed) ----
        #pragma unroll
        for (int j = 0; j < 4; ++j) {
            const _Float16* vrow = vb + (size_t)(j * 16 + lanelo) * S_ + kt;
            const v16h v0 = *(const v16h*)(vrow + 16 * lanehi);
            const v16h v1 = *(const v16h*)(vrow + 32 + 16 * lanehi);
            o[j] = __builtin_amdgcn_wmma_f32_16x16x32_f16(false, pa0, false, v0, (short)0, o[j], false, false);
            o[j] = __builtin_amdgcn_wmma_f32_16x16x32_f16(false, pa1, false, v1, (short)0, o[j], false, false);
        }
        asm volatile("s_wait_dscnt 0x0" ::: "memory");
    }

    // ---- normalize by WMMA-computed row sums; store [b][s][h*64+dk] f16 ----
    const int b = bh >> 4, h = bh & 15;
    #pragma unroll
    for (int j = 0; j < 4; ++j) {
        #pragma unroll
        for (int r = 0; r < 8; ++r) {
            const int m   = q0 + r + 8 * lanehi;
            const int col = h * 64 + j * 16 + lanelo;
            O[(size_t)(b * S_ + m) * D_ + col] = (_Float16)(o[j][r] / osum[r]);
        }
    }
}

// ---------------------------------------------------------------------------
extern "C" void kernel_launch(void* const* d_in, const int* in_sizes, int n_in,
                              void* d_out, int out_size, void* d_ws, size_t ws_size,
                              hipStream_t stream) {
    const float* q  = (const float*)d_in[0];
    const float* k  = (const float*)d_in[1];
    const float* v  = (const float*)d_in[2];
    const float* Wq = (const float*)d_in[3];
    const float* bq = (const float*)d_in[4];
    const float* Wk = (const float*)d_in[5];
    const float* bk = (const float*)d_in[6];
    const float* Wv = (const float*)d_in[7];
    const float* bv = (const float*)d_in[8];
    const float* Wo = (const float*)d_in[9];
    const float* bo = (const float*)d_in[10];

    const size_t elems = (size_t)B_ * S_ * D_;    // 4M halfs = 8MB each
    _Float16* qhb = (_Float16*)d_ws;
    _Float16* khb = qhb + elems;
    _Float16* vTb = khb + elems;
    _Float16* Ob  = vTb + elems;
    _Float16* x16 = Ob  + elems;                  // reusable f16 activation copy
    _Float16* w16 = x16 + elems;                  // reusable f16 weight copy
                                                  // total: 42MB of d_ws

    const dim3 blk(256);
    const dim3 gG(32, 16);                        // 4096/128 x 1024/64
    const dim3 gA(16, 32);                        // 2048/128 x B*H
    const dim3 gX(2048);                          // 4M elems / 2048
    const dim3 gW(512);                           // 1M elems / 2048

    cvt_f32_f16<<<gX, blk, 0, stream>>>(q,  x16);
    cvt_f32_f16<<<gW, blk, 0, stream>>>(Wq, w16);
    gemm_wmma<0><<<gG, blk, 0, stream>>>(x16, w16, bq, qhb);

    cvt_f32_f16<<<gX, blk, 0, stream>>>(k,  x16);
    cvt_f32_f16<<<gW, blk, 0, stream>>>(Wk, w16);
    gemm_wmma<0><<<gG, blk, 0, stream>>>(x16, w16, bk, khb);

    cvt_f32_f16<<<gX, blk, 0, stream>>>(v,  x16);
    cvt_f32_f16<<<gW, blk, 0, stream>>>(Wv, w16);
    gemm_wmma<1><<<gG, blk, 0, stream>>>(x16, w16, bv, vTb);

    attn_wmma<<<gA, blk, 0, stream>>>(qhb, khb, vTb, Ob);

    cvt_f32_f16<<<gW, blk, 0, stream>>>(Wo, w16);
    gemm_wmma<2><<<gG, blk, 0, stream>>>(Ob, w16, bo, (float*)d_out);
}
